// Model_3D_7335804142004
// MI455X (gfx1250) — compile-verified, matrix operand order checked by reading, not run on hardware
//
#include <hip/hip_runtime.h>
#include <hip/hip_bf16.h>

typedef __attribute__((ext_vector_type(16))) _Float16 v16h;
typedef __attribute__((ext_vector_type(8)))  _Float16 v8h;
typedef __attribute__((ext_vector_type(8)))  float    v8f;

#define EPSV 1e-5f
#define HID 256
#define BATCH 16
#define TSTEP 10
#define NBANDS 16
#define KSEL 7
#define WMAT (1024*256)              // one 1024x256 LSTM weight matrix (2^18)
#define FC1OFF (8*WMAT)              // fc1 (64x256) f16 offset in Wpack
#define FC2OFF (8*WMAT + 64*HID)     // fc2 (16x256) f16 offset in Wpack
#define WTOT  (8*WMAT + 64*HID + 16*HID)

union FragU { v16h v; v8h h[2]; };

// Build a 16x32 f16 WMMA A/B fragment for one lane.
// 16-bit layout (ISA 7.12.2): lanes 0-15 -> K {kb..kb+7, kb+16..kb+23},
// lanes 16-31 -> K {kb+8..kb+15, kb+24..kb+31}; row = lane&15 (M for A, N for B).
__device__ __forceinline__ v16h frag_ld(const _Float16* rowbase, int kb, int hi) {
    FragU f;
    f.h[0] = *(const v8h*)(rowbase + kb + hi * 8);
    f.h[1] = *(const v8h*)(rowbase + kb + 16 + hi * 8);
    return f.v;
}

// Opaque zero: defeats LICM (addresses become loop-variant, so weight loads
// are re-issued per timestep instead of hoisted + spilled to scratch) while
// keeping the base pointers' provenance intact, so InferAddressSpaces still
// promotes the loads to global_load_b128 (SADDR form) instead of flat_load.
__device__ __forceinline__ int opq_zero() {
    int z = 0;
    asm volatile("" : "+s"(z));
    return z;
}

__device__ __forceinline__ float sigf(float x) { return 1.0f / (1.0f + expf(-x)); }

// One LSTM cell for the whole (16 x 256) batch, cooperatively across 16 waves.
// Wave w owns hidden columns [16w, 16w+16) of all 4 gates; cell state c stays
// in 8 f32 VGPRs (WMMA C/D fragment layout). h is exchanged through LDS f16.
__device__ __forceinline__ void lstm_cell(
    int lane, int n0,
    const _Float16* __restrict__ xin,   // LDS 16x256 f16 (input activations)
    _Float16* __restrict__ hbuf,        // LDS 16x256 f16 (h_prev in, h_new out)
    const _Float16* __restrict__ Wih,   // global 1024x256 f16 (row = gate*256+n)
    const _Float16* __restrict__ Whh,   // global 1024x256 f16
    const float* __restrict__ bih, const float* __restrict__ bhh,
    v8f& cst)
{
    const int col = lane & 15;
    const int hi  = lane >> 4;
    const _Float16* arow = xin  + col * HID;   // A-fragment row (m = lane&15)
    const _Float16* hrow = hbuf + col * HID;
    const _Float16* wi[4]; const _Float16* wh[4];
    #pragma unroll
    for (int g = 0; g < 4; ++g) {
        wi[g] = Wih + (g * HID + n0 + col) * HID;
        wh[g] = Whh + (g * HID + n0 + col) * HID;
    }

    v8f acc[4] = {v8f{}, v8f{}, v8f{}, v8f{}};
    #pragma unroll 2
    for (int kc = 0; kc < 8; ++kc) {               // xin @ Wih^T, K = 256
        const int kb = kc * 32;
        v16h av = frag_ld(arow, kb, hi);
        #pragma unroll
        for (int g = 0; g < 4; ++g) {
            __builtin_prefetch(wi[g] + kb + 64, 0, 3);   // next K chunk
            v16h bv = frag_ld(wi[g], kb, hi);
            acc[g] = __builtin_amdgcn_wmma_f32_16x16x32_f16(false, av, false, bv,
                                                            (short)0, acc[g], false, false);
        }
    }
    #pragma unroll 2
    for (int kc = 0; kc < 8; ++kc) {               // h_prev @ Whh^T, K = 256
        const int kb = kc * 32;
        v16h av = frag_ld(hrow, kb, hi);
        #pragma unroll
        for (int g = 0; g < 4; ++g) {
            __builtin_prefetch(wh[g] + kb + 64, 0, 3);
            v16h bv = frag_ld(wh[g], kb, hi);
            acc[g] = __builtin_amdgcn_wmma_f32_16x16x32_f16(false, av, false, bv,
                                                            (short)0, acc[g], false, false);
        }
    }

    v8f hn;
    #pragma unroll
    for (int r = 0; r < 8; ++r) {
        float iv = sigf(acc[0][r] + bih[0 * HID + n0 + col] + bhh[0 * HID + n0 + col]);
        float fv = sigf(acc[1][r] + bih[1 * HID + n0 + col] + bhh[1 * HID + n0 + col]);
        float gv = tanhf(acc[2][r] + bih[2 * HID + n0 + col] + bhh[2 * HID + n0 + col]);
        float ov = sigf(acc[3][r] + bih[3 * HID + n0 + col] + bhh[3 * HID + n0 + col]);
        float cn = fv * cst[r] + iv * gv;
        cst[r] = cn;
        hn[r]  = ov * tanhf(cn);
    }
    __syncthreads();                               // all reads of h_prev done
    #pragma unroll
    for (int r = 0; r < 8; ++r) {
        int m = r + hi * 8;                        // C/D layout: VGPR r -> row
        hbuf[m * HID + n0 + col] = (_Float16)hn[r];
    }
    __syncthreads();
}

// Convert all matmul weights (f32 -> f16) into workspace:
// [A_ih, A_hh, B_ih, B_hh, C_ih, C_hh, D_ih, D_hh, fc1, fc2]
__global__ void prep_weights(const float* __restrict__ l1_wih,
                             const float* __restrict__ l1_whh,
                             const float* __restrict__ l2_wih,
                             const float* __restrict__ l2_whh,
                             const float* __restrict__ fc1_w,
                             const float* __restrict__ fc2_w,
                             _Float16* __restrict__ w)
{
    int i = blockIdx.x * blockDim.x + threadIdx.x;
    if (i >= WTOT) return;
    float v;
    if (i < 8 * WMAT) {
        int chunk  = i >> 18;                 // WMAT = 2^18
        int within = i & (WMAT - 1);
        int layer  = (chunk >> 1) & 1;
        int lstm2  = chunk >> 2;
        bool whh   = chunk & 1;
        const float* base =
            (lstm2 ? (whh ? l2_whh : l2_wih) : (whh ? l1_whh : l1_wih)) + layer * WMAT;
        v = base[within];
    } else if (i < FC2OFF) {
        v = fc1_w[i - FC1OFF];
    } else {
        v = fc2_w[i - FC2OFF];
    }
    w[i] = (_Float16)v;
}

__global__ void __launch_bounds__(512)
fused_model(const float* __restrict__ x,
            const float* __restrict__ bn0_g, const float* __restrict__ bn0_b,
            const float* __restrict__ bn0_m, const float* __restrict__ bn0_v,
            const float* __restrict__ bn1_g, const float* __restrict__ bn1_b,
            const float* __restrict__ bn1_m, const float* __restrict__ bn1_v,
            const float* __restrict__ bn2_g, const float* __restrict__ bn2_b,
            const float* __restrict__ bn2_m, const float* __restrict__ bn2_v,
            const float* __restrict__ conv1_w, const float* __restrict__ conv1_b,
            const float* __restrict__ conv2_w, const float* __restrict__ conv2_b,
            const _Float16* __restrict__ Wpack,
            const float* __restrict__ l1_bih, const float* __restrict__ l1_bhh,
            const float* __restrict__ l2_bih, const float* __restrict__ l2_bhh,
            const float* __restrict__ fc1_b, const float* __restrict__ fc2_b,
            float* __restrict__ out)
{
    __shared__ float                 y0[BATCH * 2 * NBANDS];     // bn0 output
    __shared__ alignas(16) _Float16  a1[BATCH * 64 * 6];         // conv1+bn1+relu
    __shared__ alignas(16) _Float16  feat[BATCH * HID];          // CNN features
    __shared__ alignas(16) _Float16  hA[BATCH * HID];
    __shared__ alignas(16) _Float16  hB[BATCH * HID];
    __shared__ alignas(16) _Float16  hC[BATCH * HID];
    __shared__ alignas(16) _Float16  hD[BATCH * HID];
    __shared__ float                 yg[BATCH * NBANDS];
    __shared__ float                 maskL[BATCH * NBANDS];

    const int tid  = threadIdx.x;
    const int lane = tid & 31;
    const int wave = tid >> 5;      // 16 waves
    const int n0   = wave * 16;     // hidden-unit range owned by this wave
    const int col  = lane & 15;
    const int hi   = lane >> 4;

    for (int i = tid; i < BATCH * HID; i += 512) {
        hA[i] = (_Float16)0.f; hB[i] = (_Float16)0.f;
        hC[i] = (_Float16)0.f; hD[i] = (_Float16)0.f;
    }
    v8f cA = {}, cB = {}, cC = {}, cD = {};
    __syncthreads();

    for (int t = 0; t < TSTEP; ++t) {
        // Loop-variant zero offset: keeps weight/bias loads inside the t-loop
        // (no hoist/spill) while preserving global address-space inference.
        const int z = opq_zero();
        const _Float16* Wp  = Wpack  + z;
        const float*    b1i = l1_bih + z;
        const float*    b1h = l1_bhh + z;
        const float*    b2i = l2_bih + z;
        const float*    b2h = l2_bhh + z;

        // ---- stage 1: mask + bn0 (one element per thread: tid = b*32 + c*16 + nb)
        {
            int b = tid >> 5, c = (tid >> 4) & 1, nb = tid & 15;
            float xv = x[((b * 2 + c) * TSTEP + t) * NBANDS + nb];
            if (t > 0) xv *= maskL[b * NBANDS + nb];
            float s = bn0_g[c] * rsqrtf(bn0_v[c] + EPSV);
            y0[tid] = (xv - bn0_m[c]) * s + bn0_b[c];
        }
        __syncthreads();

        // ---- stage 2: conv1 (2->64, k3, s3, pad1) + bn1 + relu
        for (int idx = tid; idx < BATCH * 64 * 6; idx += 512) {
            int b = idx / 384, r = idx - b * 384, o = r / 6, p = r - o * 6;
            float acc = conv1_b[o];
            #pragma unroll
            for (int c = 0; c < 2; ++c)
                #pragma unroll
                for (int k = 0; k < 3; ++k) {
                    int pos = 3 * p + k - 1;
                    if (pos >= 0 && pos < 16)
                        acc += y0[(b * 2 + c) * 16 + pos] * conv1_w[(o * 2 + c) * 3 + k];
                }
            float s = bn1_g[o] * rsqrtf(bn1_v[o] + EPSV);
            float v = (acc - bn1_m[o]) * s + bn1_b[o];
            a1[idx] = (_Float16)fmaxf(v, 0.f);
        }
        __syncthreads();

        // ---- stage 3: conv2 (64->256, k3, s1, pad1) + bn2 + relu + max over p
        for (int idx = tid; idx < BATCH * HID; idx += 512) {
            int b = idx >> 8, o = idx & 255;
            float s = bn2_g[o] * rsqrtf(bn2_v[o] + EPSV);
            float best = 0.f;                       // == max over p of relu(.)
            for (int p = 0; p < 6; ++p) {
                float acc = conv2_b[o];
                for (int i = 0; i < 64; ++i) {
                    const float* w = conv2_w + (o * 64 + i) * 3;
                    const _Float16* ar = a1 + (b * 64 + i) * 6;
                    #pragma unroll
                    for (int k = 0; k < 3; ++k) {
                        int pos = p + k - 1;
                        if (pos >= 0 && pos < 6) acc += (float)ar[pos] * w[k];
                    }
                }
                float v = (acc - bn2_m[o]) * s + bn2_b[o];
                best = fmaxf(best, v);
            }
            feat[idx] = (_Float16)best;
        }
        __syncthreads();

        // ---- stage 4: four LSTM cells on WMMA (all 16 waves cooperate)
        lstm_cell(lane, n0, feat, hA, Wp + 0 * WMAT, Wp + 1 * WMAT, b1i,        b1h,        cA);
        lstm_cell(lane, n0, hA,   hB, Wp + 2 * WMAT, Wp + 3 * WMAT, b1i + 1024, b1h + 1024, cB);
        lstm_cell(lane, n0, feat, hC, Wp + 4 * WMAT, Wp + 5 * WMAT, b2i,        b2h,        cC);
        lstm_cell(lane, n0, hC,   hD, Wp + 6 * WMAT, Wp + 7 * WMAT, b2i + 1024, b2h + 1024, cD);

        // ---- stage 5: FC heads on WMMA
        if (wave < 4) {                          // y_test = hB @ fc1^T  (4 N-tiles)
            const int j = wave * 16 + col;
            const _Float16* wrow = Wp + FC1OFF + j * HID;
            const _Float16* hrow = hB + col * HID;
            v8f a = {};
            #pragma unroll 2
            for (int kc = 0; kc < 8; ++kc) {
                v16h av = frag_ld(hrow, kc * 32, hi);
                v16h bv = frag_ld(wrow, kc * 32, hi);
                a = __builtin_amdgcn_wmma_f32_16x16x32_f16(false, av, false, bv,
                                                           (short)0, a, false, false);
            }
            const float bj = fc1_b[j];
            #pragma unroll
            for (int r = 0; r < 8; ++r) {
                int m = r + hi * 8;
                out[t * BATCH * 64 + m * 64 + j] = a[r] + bj;
            }
        } else if (wave == 4) {                  // y_guide = hD @ fc2^T (1 N-tile)
            const int j = col;
            const _Float16* wrow = Wp + FC2OFF + j * HID;
            const _Float16* hrow = hD + col * HID;
            v8f a = {};
            #pragma unroll 2
            for (int kc = 0; kc < 8; ++kc) {
                v16h av = frag_ld(hrow, kc * 32, hi);
                v16h bv = frag_ld(wrow, kc * 32, hi);
                a = __builtin_amdgcn_wmma_f32_16x16x32_f16(false, av, false, bv,
                                                           (short)0, a, false, false);
            }
            const float bj = fc2_b[j];
            #pragma unroll
            for (int r = 0; r < 8; ++r) {
                int m = r + hi * 8;
                float v = a[r] + bj;
                yg[m * NBANDS + j] = v;
                out[TSTEP * BATCH * 64 + t * BATCH * NBANDS + m * NBANDS + j] = v;
            }
        }
        __syncthreads();

        // ---- stage 6: argmax + top-K(7) nearest-band mask for next step
        if (tid < BATCH) {
            int b = tid;
            float bv = yg[b * NBANDS]; int m = 0;
            for (int j = 1; j < NBANDS; ++j) {
                float v = yg[b * NBANDS + j];
                if (v > bv) { bv = v; m = j; }
            }
            bool sel[NBANDS];
            #pragma unroll
            for (int j = 0; j < NBANDS; ++j) sel[j] = false;
            for (int it = 0; it < KSEL; ++it) {               // ties -> lower index
                int best = 0; float bd = -1e30f;
                for (int j = 0; j < NBANDS; ++j) {
                    if (sel[j]) continue;
                    float d = -fabsf((float)j - (float)m);
                    if (d > bd) { bd = d; best = j; }
                }
                sel[best] = true;
            }
            for (int j = 0; j < NBANDS; ++j)
                maskL[b * NBANDS + j] = sel[j] ? 1.f : 0.f;
        }
        __syncthreads();
    }
}

extern "C" void kernel_launch(void* const* d_in, const int* in_sizes, int n_in,
                              void* d_out, int out_size, void* d_ws, size_t ws_size,
                              hipStream_t stream) {
    // setup_inputs() dict order:
    const float* x       = (const float*)d_in[0];
    const float* bn0_g   = (const float*)d_in[1];
    const float* bn0_b   = (const float*)d_in[2];
    const float* bn0_m   = (const float*)d_in[3];
    const float* bn0_v   = (const float*)d_in[4];
    const float* bn1_g   = (const float*)d_in[5];
    const float* bn1_b   = (const float*)d_in[6];
    const float* bn1_m   = (const float*)d_in[7];
    const float* bn1_v   = (const float*)d_in[8];
    const float* bn2_g   = (const float*)d_in[9];
    const float* bn2_b   = (const float*)d_in[10];
    const float* bn2_m   = (const float*)d_in[11];
    const float* bn2_v   = (const float*)d_in[12];
    const float* conv1_w = (const float*)d_in[13];
    const float* conv1_b = (const float*)d_in[14];
    const float* conv2_w = (const float*)d_in[15];
    const float* conv2_b = (const float*)d_in[16];
    const float* l1_wih  = (const float*)d_in[17];
    const float* l1_whh  = (const float*)d_in[18];
    const float* l1_bih  = (const float*)d_in[19];
    const float* l1_bhh  = (const float*)d_in[20];
    const float* l2_wih  = (const float*)d_in[21];
    const float* l2_whh  = (const float*)d_in[22];
    const float* l2_bih  = (const float*)d_in[23];
    const float* l2_bhh  = (const float*)d_in[24];
    const float* fc1_w   = (const float*)d_in[25];
    const float* fc1_b   = (const float*)d_in[26];
    const float* fc2_w   = (const float*)d_in[27];
    const float* fc2_b   = (const float*)d_in[28];
    float* out = (float*)d_out;

    _Float16* Wpack = (_Float16*)d_ws;   // ~4.2 MiB f16 weights (L2-resident)

    prep_weights<<<(WTOT + 255) / 256, 256, 0, stream>>>(
        l1_wih, l1_whh, l2_wih, l2_whh, fc1_w, fc2_w, Wpack);

    fused_model<<<1, 512, 0, stream>>>(
        x, bn0_g, bn0_b, bn0_m, bn0_v,
        bn1_g, bn1_b, bn1_m, bn1_v,
        bn2_g, bn2_b, bn2_m, bn2_v,
        conv1_w, conv1_b, conv2_w, conv2_b,
        Wpack,
        l1_bih, l1_bhh, l2_bih, l2_bhh,
        fc1_b, fc2_b,
        out);
}
